// Attention_17231408791684
// MI455X (gfx1250) — matrix operand order repaired and checked by reading, not compiled, not run on hardware
//
#include <hip/hip_runtime.h>
#include <math.h>

typedef __attribute__((ext_vector_type(16))) _Float16 v16h;
typedef __attribute__((ext_vector_type(8)))  float    v8f;
typedef __attribute__((ext_vector_type(4)))  int      v4i;

union FragA { v16h v; _Float16 h[16]; unsigned u[8]; };
union FragC { v8f  v; float f[8]; };

constexpr int S_    = 2048;
constexpr int DIM_  = 2880;
constexpr int NH_   = 64;
constexpr int NKV_  = 8;
constexpr int HD_   = 64;
constexpr int QKVN_ = NH_*HD_ + 2*NKV_*HD_; // 5120

#if __has_builtin(__builtin_amdgcn_global_load_async_to_lds_b128)
#define HAVE_ASYNC_LDS 1
__device__ __forceinline__ void async_cp16(const _Float16* g, _Float16* l) {
  __builtin_amdgcn_global_load_async_to_lds_b128((v4i*)g, (v4i*)l, 0, 0);
}
#endif

#if __has_builtin(__builtin_amdgcn_s_wait_asynccnt)
#define WAIT_ASYNC(n) __builtin_amdgcn_s_wait_asynccnt(n)
#else
#define WAIT_ASYNC(n) asm volatile("s_wait_asynccnt " #n ::: "memory")
#endif

// ---------------- elementwise f32 -> f16 conversion ----------------
__global__ void cvt_f32_f16(const float* __restrict__ in, _Float16* __restrict__ out, size_t n) {
  size_t i = (size_t)blockIdx.x * blockDim.x + threadIdx.x;
  size_t stride = (size_t)gridDim.x * blockDim.x;
  for (; i < n; i += stride) out[i] = (_Float16)in[i];
}

// ---------------- f16 GEMM: C[M][N] = A[M][K] * W[N][K]^T + bias ----------------
// block = 256 threads (8 waves), tile 128x128, wave tile 32x64, K step 32,
// double-buffered LDS fed by async global->LDS DMA.
__global__ __launch_bounds__(256) void gemm_f16_wmma(
    const _Float16* __restrict__ A, const _Float16* __restrict__ W,
    const float* __restrict__ bias, float* __restrict__ C,
    int M, int N, int K) {
  __shared__ __align__(16) _Float16 As[2][128][40];
  __shared__ __align__(16) _Float16 Ws[2][128][40];
  const int tid  = threadIdx.x;
  const int lane = tid & 31;
  const int w    = tid >> 5;
  const int wm   = w & 3;      // wave M position (0..3)
  const int wn   = w >> 2;     // wave N position (0..1)
  const int bm   = blockIdx.x * 128;
  const int bn   = blockIdx.y * 128;
  const int r16  = lane & 15;
  const int hh   = lane >> 4;

  FragC acc[2][4];
  for (int mi = 0; mi < 2; mi++)
    for (int ni = 0; ni < 4; ni++)
      for (int v = 0; v < 8; v++) acc[mi][ni].f[v] = 0.0f;

  const int lrow = tid >> 1;           // 0..127
  const int lcol = (tid & 1) << 4;     // 0 or 16
  int nr = bn + lrow; if (nr > N - 1) nr = N - 1;   // clamp: OOB columns never stored
  const _Float16* Ag = A + (size_t)(bm + lrow) * K + lcol;
  const _Float16* Wg = W + (size_t)nr * K + lcol;

  const int nk = K >> 5;

#ifdef HAVE_ASYNC_LDS
  // prologue: async DMA of tile 0 into buffer 0
  async_cp16(Ag,     &As[0][lrow][lcol]);
  async_cp16(Ag + 8, &As[0][lrow][lcol + 8]);
  async_cp16(Wg,     &Ws[0][lrow][lcol]);
  async_cp16(Wg + 8, &Ws[0][lrow][lcol + 8]);
#endif

  for (int i = 0; i < nk; i++) {
    const int cur = i & 1;
#ifdef HAVE_ASYNC_LDS
    if (i + 1 < nk) {
      const int nxt = cur ^ 1;
      const _Float16* Ag2 = Ag + (size_t)(i + 1) * 32;
      const _Float16* Wg2 = Wg + (size_t)(i + 1) * 32;
      async_cp16(Ag2,     &As[nxt][lrow][lcol]);
      async_cp16(Ag2 + 8, &As[nxt][lrow][lcol + 8]);
      async_cp16(Wg2,     &Ws[nxt][lrow][lcol]);
      async_cp16(Wg2 + 8, &Ws[nxt][lrow][lcol + 8]);
      WAIT_ASYNC(4);   // our 4 in-order ops for tile i are complete
    } else {
      WAIT_ASYNC(0);
    }
    __syncthreads();   // all waves' tile-i DMA landed
#else
    {
      const uint4* ga = (const uint4*)(Ag + (size_t)i * 32);
      const uint4* gw = (const uint4*)(Wg + (size_t)i * 32);
      uint4 a0 = ga[0], a1 = ga[1];
      uint4 e0 = gw[0], e1 = gw[1];
      *(uint4*)&As[cur][lrow][lcol]     = a0;
      *(uint4*)&As[cur][lrow][lcol + 8] = a1;
      *(uint4*)&Ws[cur][lrow][lcol]     = e0;
      *(uint4*)&Ws[cur][lrow][lcol + 8] = e1;
    }
    __syncthreads();
#endif

    FragA af[2], bf[4];
    #pragma unroll
    for (int mi = 0; mi < 2; mi++) {
      const int row = wm*32 + mi*16 + r16;
      #pragma unroll
      for (int v = 0; v < 8; v++) {
        const int k = ((v >> 2) << 4) + (hh << 3) + ((v & 3) << 1); // A layout
        af[mi].u[v] = *(const unsigned*)&As[cur][row][k];
      }
    }
    #pragma unroll
    for (int ni = 0; ni < 4; ni++) {
      const int n = wn*64 + ni*16 + r16;
      #pragma unroll
      for (int v = 0; v < 8; v++) {
        const int k = ((v >> 2) << 4) + (hh << 3) + ((v & 3) << 1); // B layout
        bf[ni].u[v] = *(const unsigned*)&Ws[cur][n][k];
      }
    }
    #pragma unroll
    for (int mi = 0; mi < 2; mi++)
      #pragma unroll
      for (int ni = 0; ni < 4; ni++)
        acc[mi][ni].v = __builtin_amdgcn_wmma_f32_16x16x32_f16(
            false, af[mi].v, false, bf[ni].v, (short)0, acc[mi][ni].v, false, false);

    __syncthreads();   // safe to overwrite this buffer two iterations later
  }

  for (int mi = 0; mi < 2; mi++)
    for (int ni = 0; ni < 4; ni++) {
      const int gm = bm + wm*32 + mi*16;
      const int gn = bn + wn*64 + ni*16 + r16;
      if (gn < N) {
        const float b = bias ? bias[gn] : 0.0f;
        #pragma unroll
        for (int v = 0; v < 8; v++)
          C[(size_t)(gm + v + (hh << 3)) * N + gn] = acc[mi][ni].f[v] + b;
      }
    }
}

// ---------------- RoPE + split/reformat to f16 head-major ----------------
// qkv: [S][5120] f32.  qh: [NH][S][HD] f16, kh: [NKV][S][HD] f16, vt: [NKV][HD][S] f16.
__global__ void rope_split_kernel(const float* __restrict__ qkv, const float* __restrict__ rope,
                                  _Float16* __restrict__ qh, _Float16* __restrict__ kh,
                                  _Float16* __restrict__ vt) {
  const size_t idx = (size_t)blockIdx.x * blockDim.x + threadIdx.x;
  if (idx >= (size_t)S_ * QKVN_) return;
  const int t   = (int)(idx / QKVN_);
  const int col = (int)(idx % QKVN_);
  const float val = qkv[idx];
  if (col < NH_*HD_) {
    const int h = col >> 6, d = col & 63;
    const float c = rope[t*128 + d];
    const float s = rope[t*128 + 64 + d];
    const float other = qkv[(size_t)t * QKVN_ + (h << 6) + ((d < 32) ? (d + 32) : (d - 32))];
    const float rot = (d < 32) ? -other : other;
    qh[((size_t)h * S_ + t) * HD_ + d] = (_Float16)(val * c + rot * s);
  } else if (col < NH_*HD_ + NKV_*HD_) {
    const int cc = col - NH_*HD_;
    const int h = cc >> 6, d = cc & 63;
    const float c = rope[t*128 + d];
    const float s = rope[t*128 + 64 + d];
    const float other = qkv[(size_t)t * QKVN_ + NH_*HD_ + (h << 6) + ((d < 32) ? (d + 32) : (d - 32))];
    const float rot = (d < 32) ? -other : other;
    kh[((size_t)h * S_ + t) * HD_ + d] = (_Float16)(val * c + rot * s);
  } else {
    const int cc = col - (NH_*HD_ + NKV_*HD_);
    const int h = cc >> 6, d = cc & 63;
    vt[((size_t)(h * HD_ + d)) * S_ + t] = (_Float16)val; // transposed: key contiguous
  }
}

// ---------------- Flash attention w/ sliding window + sinks (WMMA) ----------------
// grid = (S/64, NH), block = 128 (4 waves). Each wave owns a 16-row query block.
__global__ __launch_bounds__(128) void attn_kernel(
    const _Float16* __restrict__ qh, const _Float16* __restrict__ kh,
    const _Float16* __restrict__ vt, const float* __restrict__ sinks,
    const int* __restrict__ swin, _Float16* __restrict__ ah) {
  __shared__ __align__(16) _Float16 Ps[4][16][40];
  const int tid  = threadIdx.x;
  const int lane = tid & 31, w = tid >> 5;
  const int r16  = lane & 15, hh = lane >> 4;
  const int h    = blockIdx.y;
  const int kvh  = h >> 3;
  const int q0   = blockIdx.x * 64 + w * 16;
  const int window = swin[0];
  const float scale = (0.1f * logf(32.0f) + 1.0f) * 0.125f; // mscale / sqrt(64)

  const _Float16* Qp = qh + ((size_t)h * S_ + q0) * HD_;
  const _Float16* Kp = kh + (size_t)kvh * S_ * HD_;
  const _Float16* Vp = vt + (size_t)kvh * HD_ * S_;

  FragA qa[2];
  #pragma unroll
  for (int c = 0; c < 2; c++)
    #pragma unroll
    for (int v = 0; v < 8; v++) {
      const int k = c*32 + ((v >> 2) << 4) + (hh << 3) + ((v & 3) << 1);
      qa[c].u[v] = *(const unsigned*)&Qp[(size_t)r16 * HD_ + k];
    }

  float mrow[8], lrow[8];
  FragC oacc[4];
  #pragma unroll
  for (int v = 0; v < 8; v++) { mrow[v] = -1e30f; lrow[v] = 0.0f; }
  for (int c = 0; c < 4; c++)
    for (int v = 0; v < 8; v++) oacc[c].f[v] = 0.0f;

  int js = q0 - window + 1; if (js < 0) js = 0; js &= ~31;
  for (int kb = js; kb <= q0 + 15; kb += 32) {
    // prefetch next key/value block while this one computes
    if (kb + 32 <= q0 + 15) {
      __builtin_prefetch(&Kp[(size_t)(kb + 32) * HD_], 0, 0);
      __builtin_prefetch(&Vp[(size_t)r16 * S_ + kb + 32], 0, 0);
    }
    // K^T fragments: 2 key tiles x 2 d-chunks
    FragA kf[2][2];
    #pragma unroll
    for (int nb = 0; nb < 2; nb++)
      #pragma unroll
      for (int c = 0; c < 2; c++)
        #pragma unroll
        for (int v = 0; v < 8; v++) {
          const int k = c*32 + ((v >> 2) << 4) + (hh << 3) + ((v & 3) << 1);
          kf[nb][c].u[v] = *(const unsigned*)&Kp[(size_t)(kb + nb*16 + r16) * HD_ + k];
        }
    FragC sc[2];
    #pragma unroll
    for (int nb = 0; nb < 2; nb++) {
      for (int v = 0; v < 8; v++) sc[nb].f[v] = 0.0f;
      sc[nb].v = __builtin_amdgcn_wmma_f32_16x16x32_f16(false, qa[0].v, false, kf[nb][0].v, (short)0, sc[nb].v, false, false);
      sc[nb].v = __builtin_amdgcn_wmma_f32_16x16x32_f16(false, qa[1].v, false, kf[nb][1].v, (short)0, sc[nb].v, false, false);
    }
    // online softmax over the 32 new keys
    #pragma unroll
    for (int v = 0; v < 8; v++) {
      const int i = q0 + v + (hh << 3);
      const int j0 = kb + r16;
      const int j1 = kb + 16 + r16;
      float s0 = ((j0 <= i) && (i - j0 < window)) ? sc[0].f[v] * scale : -1e30f;
      float s1 = ((j1 <= i) && (i - j1 < window)) ? sc[1].f[v] * scale : -1e30f;
      float sm = fmaxf(s0, s1);
      #pragma unroll
      for (int off = 8; off > 0; off >>= 1) sm = fmaxf(sm, __shfl_xor(sm, off, 16));
      const float nm = fmaxf(mrow[v], sm);
      const float fr = expf(mrow[v] - nm);
      const float p0 = expf(s0 - nm);
      const float p1 = expf(s1 - nm);
      float rs = p0 + p1;
      #pragma unroll
      for (int off = 8; off > 0; off >>= 1) rs += __shfl_xor(rs, off, 16);
      lrow[v] = lrow[v] * fr + rs;
      mrow[v] = nm;
      #pragma unroll
      for (int c = 0; c < 4; c++) oacc[c].f[v] *= fr;
      Ps[w][v + (hh << 3)][r16]      = (_Float16)p0;
      Ps[w][v + (hh << 3)][16 + r16] = (_Float16)p1;
    }
    asm volatile("s_wait_dscnt 0" ::: "memory");
    __builtin_amdgcn_wave_barrier();
    // P fragment (C-layout -> A-layout through LDS)
    FragA pa;
    #pragma unroll
    for (int v = 0; v < 8; v++) {
      const int k = ((v >> 2) << 4) + (hh << 3) + ((v & 3) << 1);
      pa.u[v] = *(const unsigned*)&Ps[w][r16][k];
    }
    // V fragments (transposed layout: key contiguous)
    FragA vb[4];
    #pragma unroll
    for (int c = 0; c < 4; c++)
      #pragma unroll
      for (int v = 0; v < 8; v++) {
        const int d  = c*16 + r16;
        const int kk = ((v >> 2) << 4) + (hh << 3) + ((v & 3) << 1);
        vb[c].u[v] = *(const unsigned*)&Vp[(size_t)d * S_ + kb + kk];
      }
    #pragma unroll
    for (int c = 0; c < 4; c++)
      oacc[c].v = __builtin_amdgcn_wmma_f32_16x16x32_f16(false, pa.v, false, vb[c].v, (short)0, oacc[c].v, false, false);
  }

  const float sink = sinks[h];
  float fac[8];
  #pragma unroll
  for (int v = 0; v < 8; v++) {
    const float lse = mrow[v] + logf(lrow[v]);
    const float ss  = 1.0f / (1.0f + expf(-(lse - sink)));
    fac[v] = ss / lrow[v];
  }
  #pragma unroll
  for (int c = 0; c < 4; c++)
    #pragma unroll
    for (int v = 0; v < 8; v++) {
      const int t = q0 + v + (hh << 3);
      const int d = c*16 + r16;
      ah[(size_t)t * (NH_*HD_) + h*HD_ + d] = (_Float16)(oacc[c].f[v] * fac[v]);
    }
}

// ---------------- launcher ----------------
extern "C" void kernel_launch(void* const* d_in, const int* in_sizes, int n_in,
                              void* d_out, int out_size, void* d_ws, size_t ws_size,
                              hipStream_t stream) {
  const float* x     = (const float*)d_in[0];
  const float* rope  = (const float*)d_in[1];
  const float* wq_w  = (const float*)d_in[2];
  const float* wq_b  = (const float*)d_in[3];
  const float* wk_w  = (const float*)d_in[4];
  const float* wk_b  = (const float*)d_in[5];
  const float* wv_w  = (const float*)d_in[6];
  const float* wv_b  = (const float*)d_in[7];
  const float* wo_w  = (const float*)d_in[8];
  const float* wo_b  = (const float*)d_in[9];
  const float* sinks = (const float*)d_in[10];
  const int*   swin  = (const int*)d_in[11];
  float* out = (float*)d_out;
  (void)in_sizes; (void)n_in; (void)out_size; (void)ws_size;

  char* ws = (char*)d_ws;
  size_t off = 0;
  auto alloc = [&](size_t bytes) -> void* {
    void* p = ws + off;
    off = (off + bytes + 255) & ~(size_t)255;
    return p;
  };
  _Float16* xh    = (_Float16*)alloc((size_t)S_ * DIM_ * 2);
  _Float16* wqkvh = (_Float16*)alloc((size_t)QKVN_ * DIM_ * 2);
  _Float16* woh   = (_Float16*)alloc((size_t)DIM_ * NH_ * HD_ * 2);
  float*    qkvb  = (float*)   alloc((size_t)QKVN_ * 4);
  float*    qkv   = (float*)   alloc((size_t)S_ * QKVN_ * 4);
  _Float16* qh    = (_Float16*)alloc((size_t)NH_ * S_ * HD_ * 2);
  _Float16* kh    = (_Float16*)alloc((size_t)NKV_ * S_ * HD_ * 2);
  _Float16* vt    = (_Float16*)alloc((size_t)NKV_ * HD_ * S_ * 2);
  _Float16* ah    = (_Float16*)alloc((size_t)S_ * NH_ * HD_ * 2);

  // concat QKV biases into one fp32 vector [5120]
  (void)hipMemcpyAsync(qkvb,                      wq_b, (size_t)NH_*HD_*4,  hipMemcpyDeviceToDevice, stream);
  (void)hipMemcpyAsync(qkvb + NH_*HD_,            wk_b, (size_t)NKV_*HD_*4, hipMemcpyDeviceToDevice, stream);
  (void)hipMemcpyAsync(qkvb + NH_*HD_ + NKV_*HD_, wv_b, (size_t)NKV_*HD_*4, hipMemcpyDeviceToDevice, stream);

  auto launch_cvt = [&](const float* src, _Float16* dst, size_t n) {
    int blocks = (int)((n + 256*8 - 1) / (256*8));
    cvt_f32_f16<<<blocks, 256, 0, stream>>>(src, dst, n);
  };
  launch_cvt(x,    xh,    (size_t)S_ * DIM_);
  launch_cvt(wq_w, wqkvh,                                   (size_t)NH_*HD_*DIM_);
  launch_cvt(wk_w, wqkvh + (size_t)NH_*HD_*DIM_,            (size_t)NKV_*HD_*DIM_);
  launch_cvt(wv_w, wqkvh + (size_t)(NH_*HD_+NKV_*HD_)*DIM_, (size_t)NKV_*HD_*DIM_);
  launch_cvt(wo_w, woh,  (size_t)DIM_ * NH_ * HD_);

  // fused QKV projection: [2048][5120] = xh @ wqkvh^T + qkvb
  dim3 g1(S_/128, QKVN_/128);
  gemm_f16_wmma<<<g1, 256, 0, stream>>>(xh, wqkvh, qkvb, qkv, S_, QKVN_, DIM_);

  // RoPE + split to head-major f16 (V transposed for WMMA B-layout)
  {
    size_t n = (size_t)S_ * QKVN_;
    int blocks = (int)((n + 255) / 256);
    rope_split_kernel<<<blocks, 256, 0, stream>>>(qkv, rope, qh, kh, vt);
  }

  // sliding-window attention with sinks
  dim3 g2(S_/64, NH_);
  attn_kernel<<<g2, 128, 0, stream>>>(qh, kh, vt, sinks, swin, ah);

  // output projection: out[2048][2880] = ah @ woh^T + wo_b
  dim3 g3(S_/128, (DIM_ + 127)/128);
  gemm_f16_wmma<<<g3, 256, 0, stream>>>(ah, woh, wo_b, out, S_, DIM_, NH_*HD_);
}
